// LSTMDualAttention_21921513078987
// MI455X (gfx1250) — compile-verified, matrix operand order checked by reading, not run on hardware
//
#include <hip/hip_runtime.h>
#include <hip/hip_bf16.h>

// ---------------------------------------------------------------------------
// LSTM + dual attention for MI455X (gfx1250, wave32, WMMA).
//
// Shapes: B=64, T=128, L=256, D_IN=H=ENC_H=1024, F_REP=256.
// All GEMMs run as bf16 WMMA (v_wmma_f32_16x16x32_bf16) with f32 accum.
// Weights are converted once to bf16 and stored transposed [N,K] so that the
// B-matrix fragment load pattern equals the A-matrix pattern (contiguous
// 16-byte chunks per lane, per the CDNA5 16-bit A/B VGPR layout).
//
// Roofline: all weights (~22 MB bf16) + ctx (64 MB) stay resident in the
// 192 MB L2, so the 128-step recurrence is L2/compute bound; the bf16 WMMA
// path with fused cell/attention epilogues is the right operating point.
// ---------------------------------------------------------------------------

#define Bn 64
#define Tn 128
#define Ln 256
#define Dn 1024
#define Hn 1024
#define En 1024
#define Fn 256

typedef __attribute__((ext_vector_type(16))) __bf16 v16bf;
typedef __attribute__((ext_vector_type(8)))  __bf16 v8bf;
typedef __attribute__((ext_vector_type(8)))  float  v8f;

static __device__ __forceinline__ float sigmoidf_(float x) {
    return 1.0f / (1.0f + __expf(-x));
}

// Load a 16x32 bf16 fragment (A or transposed-B) from row-major memory.
// CDNA5 16-bit A layout: lanes 0-15 hold rows 0-15 with K = 0..7 (v0..3) and
// K = 16..23 (v4..7); lanes 16-31 hold K = 8..15 and 24..31. Two b128 loads.
static __device__ __forceinline__ v16bf load_frag(const __bf16* base, int ld,
                                                  int row0, int k0, int lane) {
    int r  = row0 + (lane & 15);
    int kb = k0 + ((lane >> 4) << 3);
    const __bf16* p = base + (size_t)r * ld + kb;
    v8bf lo = *(const v8bf*)(p);
    v8bf hi = *(const v8bf*)(p + 16);
    v16bf f;
#pragma unroll
    for (int i = 0; i < 8; ++i) { f[i] = lo[i]; f[i + 8] = hi[i]; }
    return f;
}

static __device__ __forceinline__ v8f wmma_bf16(v16bf a, v16bf b, v8f c) {
    return __builtin_amdgcn_wmma_f32_16x16x32_bf16(
        /*neg_a=*/false, a, /*neg_b=*/false, b,
        /*c_mod=*/(short)0, c, /*reuse_a=*/false, /*reuse_b=*/false);
}

// ---------------------------------------------------------------------------
// Conversion kernels (one-time)
// ---------------------------------------------------------------------------
__global__ void cvt_bf16_kernel(const float* __restrict__ src,
                                __bf16* __restrict__ dst, int n) {
    int i = blockIdx.x * 256 + threadIdx.x;
    if (i < n) dst[i] = (__bf16)src[i];
}

// src[K][N] f32  ->  dst[N][K] bf16
__global__ void cvt_tr_kernel(const float* __restrict__ src,
                              __bf16* __restrict__ dst, int K, int N) {
    size_t i = (size_t)blockIdx.x * 256 + threadIdx.x;
    if (i < (size_t)K * N) {
        int n = (int)(i % N);
        int k = (int)(i / N);
        dst[(size_t)n * K + k] = (__bf16)src[i];
    }
}

// ---------------------------------------------------------------------------
// gi = x @ W_ih + b_ih    (M=8192, N=4096, K=1024), f32 out.
// Each wave computes a 16x64 strip (4 N-tiles) to reuse the A fragment.
// ---------------------------------------------------------------------------
__global__ void gemm_gi_kernel(const __bf16* __restrict__ A,
                               const __bf16* __restrict__ Bt,
                               const float* __restrict__ bias,
                               float* __restrict__ C) {
    const int K = Dn, N = 4 * Hn;
    int w    = blockIdx.x * 8 + (threadIdx.x >> 5);
    int lane = threadIdx.x & 31;
    int nb64 = N / 64;                 // 64
    int mt   = w / nb64;
    int nb   = w % nb64;
    int m0 = mt * 16, n0 = nb * 64;

    v8f acc[4] = {};
    for (int k0 = 0; k0 < K; k0 += 32) {
        // prefetch next A k-chunk (emits global_prefetch_b8 on gfx1250)
        __builtin_prefetch(A + (size_t)(m0 + (lane & 15)) * K + k0 + 32, 0, 1);
        v16bf a = load_frag(A, K, m0, k0, lane);
#pragma unroll
        for (int j = 0; j < 4; ++j) {
            v16bf b = load_frag(Bt, K, n0 + j * 16, k0, lane);
            acc[j] = wmma_bf16(a, b, acc[j]);
        }
    }
    int cl = lane & 15;
    int rb = m0 + ((lane >> 4) << 3);
#pragma unroll
    for (int j = 0; j < 4; ++j) {
        int col = n0 + j * 16 + cl;
        float bb = bias[col];
#pragma unroll
        for (int v = 0; v < 8; ++v)
            C[(size_t)(rb + v) * N + col] = acc[j][v] + bb;
    }
}

// ---------------------------------------------------------------------------
// Fused gates GEMM + LSTM cell update.
// Each wave owns one 16x16 column slice j of H and accumulates the matching
// slice of all four gate blocks (j, j+H, j+2H, j+3H), then does the cell
// nonlinearity entirely in registers.
// ---------------------------------------------------------------------------
__global__ void gates_cell_kernel(const __bf16* __restrict__ h_bf,   // [B,H] (h_tilde)
                                  const __bf16* __restrict__ Wt_hh,  // [4H,H]
                                  const float* __restrict__ gi,      // [B*T,4H]
                                  const float* __restrict__ b_hh,
                                  float* __restrict__ c_st,          // [B,H]
                                  __bf16* __restrict__ hy_bf,        // [B,H]
                                  __bf16* __restrict__ catb,         // [B,2H]
                                  float* __restrict__ c_last,        // d_out
                                  int t, int is_last) {
    const int K = Hn;
    int w    = blockIdx.x * 8 + (threadIdx.x >> 5);
    int lane = threadIdx.x & 31;
    int mt = w / (Hn / 16);            // 0..3
    int nt = w % (Hn / 16);            // 0..63
    int m0 = mt * 16, n0 = nt * 16;

    v8f ai = {}, af = {}, ag = {}, ao = {};
    for (int k0 = 0; k0 < K; k0 += 32) {
        v16bf a = load_frag(h_bf, K, m0, k0, lane);
        v16bf b0 = load_frag(Wt_hh, K, n0,          k0, lane);
        ai = wmma_bf16(a, b0, ai);
        v16bf b1 = load_frag(Wt_hh, K, n0 + Hn,     k0, lane);
        af = wmma_bf16(a, b1, af);
        v16bf b2 = load_frag(Wt_hh, K, n0 + 2 * Hn, k0, lane);
        ag = wmma_bf16(a, b2, ag);
        v16bf b3 = load_frag(Wt_hh, K, n0 + 3 * Hn, k0, lane);
        ao = wmma_bf16(a, b3, ao);
    }
    int col = n0 + (lane & 15);
    int rb  = m0 + ((lane >> 4) << 3);
    float bhi = b_hh[col], bhf = b_hh[col + Hn],
          bhg = b_hh[col + 2 * Hn], bho = b_hh[col + 3 * Hn];
#pragma unroll
    for (int v = 0; v < 8; ++v) {
        int b = rb + v;                                // batch index
        size_t g = ((size_t)b * Tn + t) * (4 * Hn) + col;
        float xi = ai[v] + gi[g]          + bhi;
        float xf = af[v] + gi[g + Hn]     + bhf;
        float xg = ag[v] + gi[g + 2 * Hn] + bhg;
        float xo = ao[v] + gi[g + 3 * Hn] + bho;
        float cp = c_st[(size_t)b * Hn + col];
        float cy = sigmoidf_(xf) * cp + sigmoidf_(xi) * tanhf(xg);
        float hy = sigmoidf_(xo) * tanhf(cy);
        c_st[(size_t)b * Hn + col]          = cy;
        hy_bf[(size_t)b * Hn + col]         = (__bf16)hy;
        catb[(size_t)b * 2 * Hn + Hn + col] = (__bf16)hy;  // [c_t, hy] 2nd half
        if (is_last) c_last[(size_t)b * Hn + col] = cy;
    }
}

// ---------------------------------------------------------------------------
// Generic M=64 GEMM: C = A @ Wt^T + bias, f32 out (used for q_c, q_f).
// A row stride must equal K. One wave per 16x16 tile.
// ---------------------------------------------------------------------------
__global__ void gemm_bias_kernel(const __bf16* __restrict__ A,
                                 const __bf16* __restrict__ Bt,
                                 const float* __restrict__ bias,
                                 float* __restrict__ C,
                                 int M, int N, int K) {
    int w    = blockIdx.x * 8 + (threadIdx.x >> 5);
    int lane = threadIdx.x & 31;
    int ntiles = N / 16;
    if (w >= (M / 16) * ntiles) return;          // wave-uniform
    int m0 = (w / ntiles) * 16;
    int n0 = (w % ntiles) * 16;
    v8f acc = {};
    for (int k0 = 0; k0 < K; k0 += 32) {
        v16bf a = load_frag(A,  K, m0, k0, lane);
        v16bf b = load_frag(Bt, K, n0, k0, lane);
        acc = wmma_bf16(a, b, acc);
    }
    int col = n0 + (lane & 15);
    int rb  = m0 + ((lane >> 4) << 3);
    float bb = bias[col];
#pragma unroll
    for (int v = 0; v < 8; ++v)
        C[(size_t)(rb + v) * N + col] = acc[v] + bb;
}

// ---------------------------------------------------------------------------
// Dual attention: scores, product-of-softmaxes (renormalized -> the two
// softmax denominators cancel: alpha = exp(sc-mc)*exp(sf-mf) / sum),
// weighted context. One workgroup (256 threads) per batch element.
// ---------------------------------------------------------------------------
__global__ void attn_kernel(const float* __restrict__ ctx,   // [B,L,E]
                            const float* __restrict__ zin,   // [B,L,F]
                            const float* __restrict__ q_c,   // [B,E]
                            const float* __restrict__ q_f,   // [B,F]
                            float* __restrict__ alpha_out,   // [T,B,L] base
                            __bf16* __restrict__ catb,       // [B,2H]
                            int t) {
    __shared__ __align__(16) float shq[En];
    __shared__ __align__(16) float shf[Fn];
    __shared__ float s1[Ln];
    __shared__ float s2[Ln];
    __shared__ float red[Ln];
    int b = blockIdx.x, tid = threadIdx.x;

    for (int i = tid; i < En; i += 256) shq[i] = q_c[(size_t)b * En + i];
    shf[tid] = q_f[(size_t)b * Fn + tid];
    __syncthreads();

    // scores: one thread per l
    {
        const float4* cr = (const float4*)(ctx + ((size_t)b * Ln + tid) * En);
        const float4* qq = (const float4*)shq;
        float ac = 0.f;
        for (int e = 0; e < En / 4; ++e) {
            float4 v = cr[e], q = qq[e];
            ac += v.x * q.x + v.y * q.y + v.z * q.z + v.w * q.w;
        }
        const float4* fr = (const float4*)(zin + ((size_t)b * Ln + tid) * Fn);
        const float4* qf = (const float4*)shf;
        float af = 0.f;
        for (int e = 0; e < Fn / 4; ++e) {
            float4 v = fr[e], q = qf[e];
            af += v.x * q.x + v.y * q.y + v.z * q.z + v.w * q.w;
        }
        s1[tid] = ac;
        s2[tid] = af;
    }
    __syncthreads();

    // max of s1
    red[tid] = s1[tid]; __syncthreads();
    for (int s = 128; s > 0; s >>= 1) {
        if (tid < s) red[tid] = fmaxf(red[tid], red[tid + s]);
        __syncthreads();
    }
    float m1 = red[0]; __syncthreads();
    // max of s2
    red[tid] = s2[tid]; __syncthreads();
    for (int s = 128; s > 0; s >>= 1) {
        if (tid < s) red[tid] = fmaxf(red[tid], red[tid + s]);
        __syncthreads();
    }
    float m2 = red[0]; __syncthreads();

    float a = __expf(s1[tid] - m1) * __expf(s2[tid] - m2);
    red[tid] = a; __syncthreads();
    for (int s = 128; s > 0; s >>= 1) {
        if (tid < s) red[tid] += red[tid + s];
        __syncthreads();
    }
    float alpha = a / red[0];
    __syncthreads();
    s1[tid] = alpha;                       // reuse s1 as alpha
    alpha_out[((size_t)t * Bn + b) * Ln + tid] = alpha;
    __syncthreads();

    // c_t[e] = sum_l alpha[l] * ctx[b,l,e]; each thread owns 4 e's
    float4 acc = {0.f, 0.f, 0.f, 0.f};
    const float* cb = ctx + (size_t)b * Ln * En + tid * 4;
    for (int l = 0; l < Ln; ++l) {
        float al = s1[l];
        float4 v = *(const float4*)(cb + (size_t)l * En);
        acc.x += al * v.x; acc.y += al * v.y;
        acc.z += al * v.z; acc.w += al * v.w;
    }
    __bf16* cc = catb + (size_t)b * 2 * Hn + tid * 4;
    cc[0] = (__bf16)acc.x; cc[1] = (__bf16)acc.y;
    cc[2] = (__bf16)acc.z; cc[3] = (__bf16)acc.w;
}

// ---------------------------------------------------------------------------
// Output projection: h_tilde = tanh([c_t, hy] @ W_out + b_out).
// Writes f32 to output[b][t][:], bf16 recurrent state, and h_last on t=T-1.
// ---------------------------------------------------------------------------
__global__ void out_kernel(const __bf16* __restrict__ catb,   // [B,2H]
                           const __bf16* __restrict__ Wt_out, // [H,2H]
                           const float* __restrict__ bias,
                           float* __restrict__ out_bTH,       // [B,T,H]
                           float* __restrict__ h_last,
                           __bf16* __restrict__ h_bf,         // [B,H]
                           int t, int is_last) {
    const int K = 2 * Hn;
    int w    = blockIdx.x * 8 + (threadIdx.x >> 5);
    int lane = threadIdx.x & 31;
    int mt = w / (Hn / 16);
    int nt = w % (Hn / 16);
    int m0 = mt * 16, n0 = nt * 16;
    v8f acc = {};
    for (int k0 = 0; k0 < K; k0 += 32) {
        v16bf a = load_frag(catb,   K, m0, k0, lane);
        v16bf b = load_frag(Wt_out, K, n0, k0, lane);
        acc = wmma_bf16(a, b, acc);
    }
    int col = n0 + (lane & 15);
    int rb  = m0 + ((lane >> 4) << 3);
    float bb = bias[col];
#pragma unroll
    for (int v = 0; v < 8; ++v) {
        int b = rb + v;
        float hval = tanhf(acc[v] + bb);
        out_bTH[((size_t)b * Tn + t) * Hn + col] = hval;
        h_bf[(size_t)b * Hn + col] = (__bf16)hval;
        if (is_last) h_last[(size_t)b * Hn + col] = hval;
    }
}

// ---------------------------------------------------------------------------
// Host launcher
// ---------------------------------------------------------------------------
extern "C" void kernel_launch(void* const* d_in, const int* in_sizes, int n_in,
                              void* d_out, int out_size, void* d_ws, size_t ws_size,
                              hipStream_t stream) {
    const float* x     = (const float*)d_in[0];
    const float* hx    = (const float*)d_in[1];
    const float* cx    = (const float*)d_in[2];
    const float* zin   = (const float*)d_in[3];
    const float* ctx   = (const float*)d_in[4];
    const float* W_ih  = (const float*)d_in[5];
    const float* b_ih  = (const float*)d_in[6];
    const float* W_hh  = (const float*)d_in[7];
    const float* b_hh  = (const float*)d_in[8];
    const float* W_a   = (const float*)d_in[9];
    const float* b_a   = (const float*)d_in[10];
    const float* W_f   = (const float*)d_in[11];
    const float* b_f   = (const float*)d_in[12];
    const float* W_out = (const float*)d_in[13];
    const float* b_out = (const float*)d_in[14];

    float* out_bTH = (float*)d_out;                        // [B,T,H]
    float* h_last  = out_bTH + (size_t)Bn * Tn * Hn;       // [B,H]
    float* c_last  = h_last + (size_t)Bn * Hn;             // [B,H]
    float* alphas  = c_last + (size_t)Bn * Hn;             // [T,B,L]

    char* ws = (char*)d_ws;
    size_t o = 0;
    __bf16* x_bf   = (__bf16*)(ws + o); o += (size_t)Bn * Tn * Dn * 2;       // 16 MB
    __bf16* Wt_ih  = (__bf16*)(ws + o); o += (size_t)4 * Hn * Dn * 2;        // 8 MB
    __bf16* Wt_hh  = (__bf16*)(ws + o); o += (size_t)4 * Hn * Hn * 2;        // 8 MB
    __bf16* Wt_a   = (__bf16*)(ws + o); o += (size_t)En * Hn * 2;            // 2 MB
    __bf16* Wt_f   = (__bf16*)(ws + o); o += (size_t)Fn * Hn * 2;            // 0.5 MB
    __bf16* Wt_out = (__bf16*)(ws + o); o += (size_t)Hn * 2 * Hn * 2;        // 4 MB
    float*  gi     = (float*)(ws + o);  o += (size_t)Bn * Tn * 4 * Hn * 4;   // 128 MB
    __bf16* h_bf   = (__bf16*)(ws + o); o += (size_t)Bn * Hn * 2;            // h_tilde state
    __bf16* hy_bf  = (__bf16*)(ws + o); o += (size_t)Bn * Hn * 2;            // per-step hy
    float*  c_st   = (float*)(ws + o);  o += (size_t)Bn * Hn * 4;
    float*  q_c    = (float*)(ws + o);  o += (size_t)Bn * En * 4;
    float*  q_f    = (float*)(ws + o);  o += (size_t)Bn * Fn * 4;
    __bf16* catb   = (__bf16*)(ws + o); o += (size_t)Bn * 2 * Hn * 2;

    // --- one-time conversions ---
    {
        int n = Bn * Tn * Dn;
        cvt_bf16_kernel<<<(n + 255) / 256, 256, 0, stream>>>(x, x_bf, n);
    }
    {
        int n = Bn * Hn;
        cvt_bf16_kernel<<<(n + 255) / 256, 256, 0, stream>>>(hx, h_bf, n);
    }
    hipMemcpyAsync(c_st, cx, (size_t)Bn * Hn * sizeof(float),
                   hipMemcpyDeviceToDevice, stream);

    cvt_tr_kernel<<<(Dn * 4 * Hn + 255) / 256, 256, 0, stream>>>(W_ih, Wt_ih, Dn, 4 * Hn);
    cvt_tr_kernel<<<(Hn * 4 * Hn + 255) / 256, 256, 0, stream>>>(W_hh, Wt_hh, Hn, 4 * Hn);
    cvt_tr_kernel<<<(Hn * En + 255) / 256, 256, 0, stream>>>(W_a, Wt_a, Hn, En);
    cvt_tr_kernel<<<(Hn * Fn + 255) / 256, 256, 0, stream>>>(W_f, Wt_f, Hn, Fn);
    cvt_tr_kernel<<<(2 * Hn * Hn + 255) / 256, 256, 0, stream>>>(W_out, Wt_out, 2 * Hn, Hn);

    // --- gi = x @ W_ih + b_ih : 32768 waves -> 4096 blocks of 8 waves ---
    {
        int waves = (Bn * Tn / 16) * (4 * Hn / 64);
        gemm_gi_kernel<<<waves / 8, 256, 0, stream>>>(x_bf, Wt_ih, b_ih, gi);
    }

    // --- sequential scan ---
    for (int t = 0; t < Tn; ++t) {
        int last = (t == Tn - 1) ? 1 : 0;
        // gates + cell: 4*64 = 256 waves (reads h_bf, writes hy_bf/catb/c_st)
        gates_cell_kernel<<<32, 256, 0, stream>>>(h_bf, Wt_hh, gi, b_hh, c_st,
                                                  hy_bf, catb, c_last, t, last);
        // q_c = hy @ W_a + b_a  (256 waves)
        gemm_bias_kernel<<<32, 256, 0, stream>>>(hy_bf, Wt_a, b_a, q_c,
                                                 Bn, En, Hn);
        // q_f = hy @ W_f + b_f  (64 waves)
        gemm_bias_kernel<<<8, 256, 0, stream>>>(hy_bf, Wt_f, b_f, q_f,
                                                Bn, Fn, Hn);
        // attention scores + softmax-product + weighted context (fills catb[:H])
        attn_kernel<<<Bn, 256, 0, stream>>>(ctx, zin, q_c, q_f, alphas, catb, t);
        // h_tilde = tanh(cat @ W_out + b_out); updates h_bf for next step
        out_kernel<<<32, 256, 0, stream>>>(catb, Wt_out, b_out, out_bTH,
                                           h_last, h_bf, t, last);
    }
    (void)in_sizes; (void)n_in; (void)out_size; (void)ws_size; (void)o;
}
